// ESSAttn_8830452760655
// MI455X (gfx1250) — matrix operand order repaired and checked by reading, not compiled, not run on hardware
//
#include <hip/hip_runtime.h>
#include <hip/hip_bf16.h>

// ESS linear-attention block for MI455X (gfx1250), wave32 + WMMA f16 (f32 acc)
// + Tensor Data Mover async staging of the x tiles (double buffered).
// pass1: QKV gemm -> row stats (fp32) -> per-block partial KV = kk^T v and colsum2
// pass2: reduce slabs, finalize kv (column norm), build WW = [kv*WlT/sqrt(N) ; WlT]
// pass3: recompute Q,V gemm -> q2n -> out = [q2n|v]*WW + b_ln, transposed store.

#define B_    8
#define C_    64
#define N_    65536
#define TILE  64
#define QLD   193          // padded LDS row stride (odd -> conflict-free column access)
#define LDX   65           // x-tile LDS row stride (TDM pad: +1 dword per 64 dwords)
#define XBUF  (C_ * LDX)   // one x staging buffer (floats)
#define SLABS 64           // blocks per batch in pass1/pass3
#define ITERS (N_ / (SLABS * TILE))   // 16

typedef __attribute__((ext_vector_type(16))) _Float16 v16h;
typedef __attribute__((ext_vector_type(8)))  float    v8f;
typedef __attribute__((ext_vector_type(4)))  unsigned int v4u;
typedef __attribute__((ext_vector_type(8)))  int      v8i;
typedef __attribute__((ext_vector_type(4)))  int      v4i;

#if defined(__AMDGCN__) && __has_builtin(__builtin_amdgcn_tensor_load_to_lds) && \
    __has_builtin(__builtin_amdgcn_s_wait_tensorcnt)
#define USE_TDM 1
#else
#define USE_TDM 0
#endif

static __device__ __forceinline__ v8f wmma_f16(v16h a, v16h b, v8f c) {
  // v_wmma_f32_16x16x32_f16 : D = A(16x32) * B(32x16) + C(16x16 f32)
  return __builtin_amdgcn_wmma_f32_16x16x32_f16(
      /*neg_a=*/false, a, /*neg_b=*/false, b,
      /*c_mod=*/(short)0, c, /*reuse_a=*/false, /*reuse_b=*/false);
}

// A-fragment K offset for element idx (16-bit A 16x32 layout, ISA 7.12.2):
// lanes 0-15: V0..3 -> K 0..7, V4..7 -> K 16..23 ; lanes 16-31: +8
static __device__ __forceinline__ int a_kof(int idx, int hi) {
  return (idx & 7) + ((idx >> 3) << 4) + 8 * hi;
}

#if USE_TDM
typedef __attribute__((address_space(3))) void lds_void;

static __device__ __forceinline__ unsigned lds_byte_off(void* p) {
  return (unsigned)(unsigned long long)(lds_void*)p;
}

// TDM: DMA one 64(row) x 64(col) f32 tile, row stride = N_ floats, into LDS with
// 1-dword padding after every 64 dwords (=> LDS row stride LDX=65 floats).
static __device__ __forceinline__ void tdm_load_x_tile(const float* gsrc, float* lds_dst) {
  const unsigned long long ga = (unsigned long long)gsrc;
  v4u g0;
  g0[0] = 1u;                                   // count=1, user descriptor
  g0[1] = lds_byte_off(lds_dst);                // lds_addr (bytes)
  g0[2] = (unsigned)ga;                         // global_addr[31:0]
  g0[3] = (unsigned)(ga >> 32) | 0x80000000u;   // global_addr[56:32] | type=2
  v8i g1;
  g1[0] = 0x01520000;            // data_size=4B, pad_enable, interval=64dw, amount=1dw
  g1[1] = 0;                     // abar=0, tensor_dim0[15:0] (65536 -> 0)
  g1[2] = (int)((64u << 16) | 1u);  // tensor_dim0[31:16]=1, tensor_dim1[15:0]=64
  g1[3] = (int)(64u << 16);         // tensor_dim1[31:16]=0, tile_dim0=64
  g1[4] = 64;                       // tile_dim1=64, tile_dim2=0
  g1[5] = (int)N_;                  // tensor_dim0_stride[31:0]
  g1[6] = 0;                        // stride hi / dim1_stride lo
  g1[7] = 0;
  v4i z4 = {0, 0, 0, 0};
#if __clang_major__ >= 23
  v8i z8 = {0, 0, 0, 0, 0, 0, 0, 0};
  __builtin_amdgcn_tensor_load_to_lds(g0, g1, z4, z4, z8, 0);
#else
  __builtin_amdgcn_tensor_load_to_lds(g0, g1, z4, z4, 0);
#endif
}
#endif // USE_TDM

// Fallback: cooperative synchronous stage (same LDS layout).
static __device__ __forceinline__ void stage_x_fallback(const float* gsrc, float* dst, int tid) {
  for (int idx = tid; idx < C_ * TILE; idx += 256) {
    const int c = idx >> 6, i = idx & 63;
    dst[c * LDX + i] = gsrc[(size_t)c * N_ + i];
  }
}

// ---------------------------------------------------------------------------
__global__ __launch_bounds__(256) void ess_pass1(
    const float* __restrict__ x, const float* __restrict__ wq,
    const float* __restrict__ bq, float* __restrict__ ws_kv,
    float* __restrict__ ws_col)
{
  __shared__ float sq[TILE * QLD];   // [i][d] : q | k | v  (49.4 KB)
  __shared__ float xs[2 * XBUF];     // double-buffered x tile (33.3 KB)
  const int tid  = threadIdx.x;
  const int lane = tid & 31;
  const int wv   = tid >> 5;
  const int m    = lane & 15;        // row/col index within 16x16 tile
  const int hi   = lane >> 4;        // lane half
  const int slab = blockIdx.x;
  const int b    = blockIdx.y;
  const float* xb = x + (size_t)b * C_ * N_;

  v8f kvacc[2];
  #pragma unroll
  for (int tt = 0; tt < 2; ++tt)
    #pragma unroll
    for (int r = 0; r < 8; ++r) kvacc[tt][r] = 0.0f;
  float colacc = 0.0f;

  // prime buffer 0
#if USE_TDM
  if (wv == 0) tdm_load_x_tile(xb + (size_t)slab * ITERS * TILE, xs);
#else
  stage_x_fallback(xb + (size_t)slab * ITERS * TILE, xs, tid);
#endif

  for (int it = 0; it < ITERS; ++it) {
    const int n0 = (slab * ITERS + it) * TILE;
    const float* xcur = xs + (it & 1) * XBUF;

    // stage next tile asynchronously, then wait for current tile
#if USE_TDM
    if (wv == 0) {
      if (it + 1 < ITERS) {
        tdm_load_x_tile(xb + (size_t)(n0 + TILE), xs + ((it + 1) & 1) * XBUF);
        __builtin_amdgcn_s_wait_tensorcnt(1);
      } else {
        __builtin_amdgcn_s_wait_tensorcnt(0);
      }
    }
#else
    if (it + 1 < ITERS)
      stage_x_fallback(xb + (size_t)(n0 + TILE), xs + ((it + 1) & 1) * XBUF, tid);
#endif
    __syncthreads();

    // ---- QKV tile: [TILE x 192] = x_tile^T * Wqkv^T + b_qkv ; 4x12 wmma tiles
    for (int t = wv; t < 48; t += 8) {
      const int mt = t / 12, nt = t % 12;
      const int d0 = nt * 16;
      const float bias = bq[d0 + m];
      v8f acc;
      #pragma unroll
      for (int r = 0; r < 8; ++r) acc[r] = bias;
      #pragma unroll
      for (int k0 = 0; k0 < C_; k0 += 32) {
        v16h af, bf;
        #pragma unroll
        for (int i = 0; i < 16; ++i) {
          // A[i][c] = x[b][c][n0+i]  (from LDS-staged tile)
          af[i] = (_Float16)xcur[(k0 + a_kof(i, hi)) * LDX + mt * 16 + m];
          // B[c][d] = wq[d][c]
          bf[i] = (_Float16)wq[(d0 + m) * C_ + k0 + a_kof(i, hi)];
        }
        acc = wmma_f16(af, bf, acc);
      }
      #pragma unroll
      for (int r = 0; r < 8; ++r)
        sq[(mt * 16 + r + 8 * hi) * QLD + d0 + m] = acc[r];
    }
    __syncthreads();

    // ---- per-row stats in fp32: center q,k ; q2n (row l2) ; kk = k2/(k2s+eps)
    if (tid < TILE) {
      float* row = sq + tid * QLD;
      float sqm = 0.f, skm = 0.f;
      for (int c = 0; c < C_; ++c) { sqm += row[c]; skm += row[C_ + c]; }
      sqm *= (1.0f / C_); skm *= (1.0f / C_);
      float q2s = 0.f, k2s = 0.f, q4s = 0.f;
      for (int c = 0; c < C_; ++c) {
        float qc = row[c] - sqm;      float q2 = qc * qc;
        float kc = row[C_ + c] - skm; float k2 = kc * kc;
        q2s += q2; k2s += k2; q4s += q2 * q2;
        row[c] = q2; row[C_ + c] = k2;
      }
      const float iq = 1.0f / (q2s + 1e-7f);
      const float ik = 1.0f / (k2s + 1e-7f);
      const float qscale = iq / fmaxf(iq * sqrtf(q4s), 1e-12f);
      for (int c = 0; c < C_; ++c) { row[c] *= qscale; row[C_ + c] *= ik; }
    }
    __syncthreads();

    // ---- colsum2 partial (for column norm of kk)
    if (tid < C_) {
      float s = 0.f;
      for (int i = 0; i < TILE; ++i) { float v = sq[i * QLD + C_ + tid]; s += v * v; }
      colacc += s;
    }

    // ---- KV partial: [64x64] += kk^T * v ; 2 tiles per wave, acc lives in VGPRs
    #pragma unroll
    for (int tt = 0; tt < 2; ++tt) {
      const int t = wv * 2 + tt;
      const int mt = t >> 2, nt = t & 3;
      v8f acc = kvacc[tt];
      #pragma unroll
      for (int k0 = 0; k0 < TILE; k0 += 32) {
        v16h af, bf;
        #pragma unroll
        for (int i = 0; i < 16; ++i) {
          // A[c][i] = kk[i][c]
          af[i] = (_Float16)sq[(k0 + a_kof(i, hi)) * QLD + C_ + mt * 16 + m];
          // B[i][d] = v[i][d]
          bf[i] = (_Float16)sq[(k0 + a_kof(i, hi)) * QLD + 2 * C_ + nt * 16 + m];
        }
        acc = wmma_f16(af, bf, acc);
      }
      kvacc[tt] = acc;
    }
    __syncthreads();
  }

  // ---- store private slab (no atomics)
  float* dst = ws_kv + (size_t)(b * SLABS + slab) * (C_ * C_);
  #pragma unroll
  for (int tt = 0; tt < 2; ++tt) {
    const int t = wv * 2 + tt;
    const int mt = t >> 2, nt = t & 3;
    #pragma unroll
    for (int r = 0; r < 8; ++r)
      dst[(mt * 16 + r + 8 * hi) * C_ + nt * 16 + m] = kvacc[tt][r];
  }
  if (tid < C_) ws_col[(b * SLABS + slab) * C_ + tid] = colacc;
}

// ---------------------------------------------------------------------------
__global__ __launch_bounds__(256) void ess_pass2(
    const float* __restrict__ ws_kv, const float* __restrict__ ws_col,
    const float* __restrict__ wl, float* __restrict__ ws_ww)
{
  __shared__ float kvs[C_ * C_];
  __shared__ float cinv[C_];
  const int b = blockIdx.x;
  const int t = threadIdx.x;

  if (t < C_) {
    float s = 0.f;
    for (int sl = 0; sl < SLABS; ++sl) s += ws_col[(b * SLABS + sl) * C_ + t];
    cinv[t] = 1.0f / fmaxf(sqrtf(s), 1e-12f);
  }
  __syncthreads();

  for (int idx = t; idx < C_ * C_; idx += 256) {
    float s = 0.f;
    for (int sl = 0; sl < SLABS; ++sl)
      s += ws_kv[(size_t)(b * SLABS + sl) * (C_ * C_) + idx];
    kvs[idx] = s * cinv[idx >> 6];           // kv[c][d], column-normalized
  }
  __syncthreads();

  const float rsN = 0.00390625f;             // 1/sqrt(65536)
  for (int idx = t; idx < C_ * C_; idx += 256) {
    const int c = idx >> 6, dp = idx & 63;
    float s = 0.f;
    for (int d = 0; d < C_; ++d) s += kvs[c * C_ + d] * wl[dp * C_ + d];
    ws_ww[((size_t)b * 2 * C_ + c) * C_ + dp] = s * rsN;        // KW rows 0..63
  }
  for (int idx = t; idx < C_ * C_; idx += 256) {
    const int d = idx >> 6, dp = idx & 63;
    ws_ww[((size_t)b * 2 * C_ + C_ + d) * C_ + dp] = wl[dp * C_ + d];  // WlT rows 64..127
  }
}

// ---------------------------------------------------------------------------
__global__ __launch_bounds__(256) void ess_pass3(
    const float* __restrict__ x, const float* __restrict__ wq,
    const float* __restrict__ bq, const float* __restrict__ ww_g,
    const float* __restrict__ bln, float* __restrict__ out)
{
  __shared__ float sq[TILE * QLD];
  __shared__ float xs[2 * XBUF];
  const int tid  = threadIdx.x;
  const int lane = tid & 31;
  const int wv   = tid >> 5;
  const int m    = lane & 15;
  const int hi   = lane >> 4;
  const int slab = blockIdx.x;
  const int b    = blockIdx.y;
  const float* xb = x + (size_t)b * C_ * N_;
  const float* ww = ww_g + (size_t)b * 2 * C_ * C_;
  float* ob = out + (size_t)b * C_ * N_;

#if USE_TDM
  if (wv == 0) tdm_load_x_tile(xb + (size_t)slab * ITERS * TILE, xs);
#else
  stage_x_fallback(xb + (size_t)slab * ITERS * TILE, xs, tid);
#endif

  for (int it = 0; it < ITERS; ++it) {
    const int n0 = (slab * ITERS + it) * TILE;
    const float* xcur = xs + (it & 1) * XBUF;

#if USE_TDM
    if (wv == 0) {
      if (it + 1 < ITERS) {
        tdm_load_x_tile(xb + (size_t)(n0 + TILE), xs + ((it + 1) & 1) * XBUF);
        __builtin_amdgcn_s_wait_tensorcnt(1);
      } else {
        __builtin_amdgcn_s_wait_tensorcnt(0);
      }
    }
#else
    if (it + 1 < ITERS)
      stage_x_fallback(xb + (size_t)(n0 + TILE), xs + ((it + 1) & 1) * XBUF, tid);
#endif
    __syncthreads();

    // ---- recompute only Q (nt 0..3) and V (nt 8..11) tiles
    for (int t = wv; t < 32; t += 8) {
      const int mt = t >> 3;
      const int ns = t & 7;
      const int nt = (ns < 4) ? ns : ns + 4;
      const int d0 = nt * 16;
      const float bias = bq[d0 + m];
      v8f acc;
      #pragma unroll
      for (int r = 0; r < 8; ++r) acc[r] = bias;
      #pragma unroll
      for (int k0 = 0; k0 < C_; k0 += 32) {
        v16h af, bf;
        #pragma unroll
        for (int i = 0; i < 16; ++i) {
          af[i] = (_Float16)xcur[(k0 + a_kof(i, hi)) * LDX + mt * 16 + m];
          bf[i] = (_Float16)wq[(d0 + m) * C_ + k0 + a_kof(i, hi)];
        }
        acc = wmma_f16(af, bf, acc);
      }
      #pragma unroll
      for (int r = 0; r < 8; ++r)
        sq[(mt * 16 + r + 8 * hi) * QLD + d0 + m] = acc[r];
    }
    __syncthreads();

    // ---- q2n rows (fp32)
    if (tid < TILE) {
      float* row = sq + tid * QLD;
      float sm = 0.f;
      for (int c = 0; c < C_; ++c) sm += row[c];
      sm *= (1.0f / C_);
      float q2s = 0.f, q4s = 0.f;
      for (int c = 0; c < C_; ++c) {
        float qc = row[c] - sm; float q2 = qc * qc;
        q2s += q2; q4s += q2 * q2;
        row[c] = q2;
      }
      const float iq = 1.0f / (q2s + 1e-7f);
      const float qscale = iq / fmaxf(iq * sqrtf(q4s), 1e-12f);
      for (int c = 0; c < C_; ++c) row[c] *= qscale;
    }
    __syncthreads();

    // ---- out tile = [q2n | v] (K=128) * WW + b_ln ; write into dead kk columns
    #pragma unroll
    for (int tt = 0; tt < 2; ++tt) {
      const int t = wv * 2 + tt;
      const int mt = t >> 2, nt = t & 3;
      const int d0 = nt * 16;
      const float bias = bln[d0 + m];
      v8f acc;
      #pragma unroll
      for (int r = 0; r < 8; ++r) acc[r] = bias;
      #pragma unroll
      for (int k0 = 0; k0 < 2 * C_; k0 += 32) {
        v16h af, bf;
        #pragma unroll
        for (int i = 0; i < 16; ++i) {
          const int ka = k0 + a_kof(i, hi);
          const int ca = (ka < C_) ? ka : ka + C_;   // q2n cols 0..63, v cols 128..191
          af[i] = (_Float16)sq[(mt * 16 + m) * QLD + ca];
          bf[i] = (_Float16)ww[ka * C_ + d0 + m];
        }
        acc = wmma_f16(af, bf, acc);
      }
      #pragma unroll
      for (int r = 0; r < 8; ++r)
        sq[(mt * 16 + r + 8 * hi) * QLD + C_ + d0 + m] = acc[r];
    }
    __syncthreads();

    // ---- transposed, coalesced store into [b, C, H, W]
    for (int idx = tid; idx < C_ * TILE; idx += 256) {
      const int i = idx & (TILE - 1);
      const int dp = idx >> 6;
      ob[(size_t)dp * N_ + n0 + i] = sq[i * QLD + C_ + dp];
    }
    __syncthreads();
  }
}

// ---------------------------------------------------------------------------
extern "C" void kernel_launch(void* const* d_in, const int* in_sizes, int n_in,
                              void* d_out, int out_size, void* d_ws, size_t ws_size,
                              hipStream_t stream) {
  const float* x   = (const float*)d_in[0];
  const float* wq  = (const float*)d_in[1];
  const float* bq  = (const float*)d_in[2];
  const float* wl  = (const float*)d_in[3];
  const float* bln = (const float*)d_in[4];
  float* out = (float*)d_out;

  float* ws     = (float*)d_ws;
  float* ws_kv  = ws;                                    // B*SLABS*64*64 = 8 MB
  float* ws_col = ws_kv + (size_t)B_ * SLABS * C_ * C_;  // B*SLABS*64
  float* ws_ww  = ws_col + (size_t)B_ * SLABS * C_;      // B*128*64
  (void)in_sizes; (void)n_in; (void)out_size; (void)ws_size;

  dim3 grid(SLABS, B_);
  ess_pass1<<<grid, 256, 0, stream>>>(x, wq, bq, ws_kv, ws_col);
  ess_pass2<<<B_, 256, 0, stream>>>(ws_kv, ws_col, wl, ws_ww);
  ess_pass3<<<grid, 256, 0, stream>>>(x, wq, bq, ws_ww, bln, out);
}
